// SEThing_12257836663299
// MI455X (gfx1250) — matrix operand order repaired and checked by reading, not compiled, not run on hardware
//
#include <hip/hip_runtime.h>
#include <hip/hip_bf16.h>
#include <math.h>

// ---------------------------------------------------------------------------
// Types for CDNA5 WMMA (wave32): v_wmma_f32_16x16x32_bf16
// ---------------------------------------------------------------------------
typedef __attribute__((ext_vector_type(16))) __bf16          v16bf;
typedef __attribute__((ext_vector_type(16))) short           v16s;
typedef __attribute__((ext_vector_type(8)))  float           v8f;
typedef __attribute__((ext_vector_type(8)))  unsigned short  u16x8;

union FragCast { v16s s; v16bf b; };

#define NEGINF (-__builtin_inff())

// Problem constants
constexpr int T_     = 256;
constexpr int BSZ    = 8;
constexpr int DIM    = 512;
constexpr int NE     = 8192;     // nelen*nne
constexpr int VOCAB  = 32000;
constexpr int SRCL   = 256;
constexpr int ROWW   = NE + VOCAB + SRCL;           // 40448 (row-major concat)
constexpr long long LBATCH = (long long)T_ * ROWW;  // 10,354,688 per batch
constexpr int CHUNK  = 16384;                       // LBATCH % CHUNK == 0
constexpr int NBLK   = (int)(LBATCH / CHUNK);       // 632

__device__ __forceinline__ unsigned short f2bf(float f) {
  unsigned int u = __float_as_uint(f);
  u += 0x7fffu + ((u >> 16) & 1u);                  // round-to-nearest-even
  return (unsigned short)(u >> 16);
}

// Raw 8-element stage register block: holds loaded bytes, converts at store.
template <bool F32> struct St8;
template <> struct St8<true> {
  float4 x0, x1;
  __device__ __forceinline__ void load(const void* p, size_t off) {
    const float* f = (const float*)p;
    x0 = *(const float4*)(f + off);
    x1 = *(const float4*)(f + off + 4);
  }
  __device__ __forceinline__ u16x8 pack() const {
    u16x8 r;
    r[0] = f2bf(x0.x); r[1] = f2bf(x0.y); r[2] = f2bf(x0.z); r[3] = f2bf(x0.w);
    r[4] = f2bf(x1.x); r[5] = f2bf(x1.y); r[6] = f2bf(x1.z); r[7] = f2bf(x1.w);
    return r;
  }
};
template <> struct St8<false> {
  u16x8 v;
  __device__ __forceinline__ void load(const void* p, size_t off) {
    v = *(const u16x8*)((const unsigned short*)p + off);
  }
  __device__ __forceinline__ u16x8 pack() const { return v; }
};

// ---------------------------------------------------------------------------
// Generic NT GEMM:  O[m,n] = sum_k A[m,k] * W[n,k]
//  - A, W row-major (f32 or bf16-as-ushort), staged to LDS as bf16
//  - RMAP (compile-time): A row r -> memory row (r%rmap_dim)*rmap_mul + r/rmap_dim
//  - batch strides (element offsets), blockIdx.z = batch
//  - block tile 128x128, 8 waves, wave tile 64x32 (8 WMMA / K-step), BK=32
//  - distance-2 software pipeline over double-buffered LDS
// ---------------------------------------------------------------------------
constexpr int BM = 128, BN = 128, BK = 32;
constexpr int AP = 40, WP = 40;   // LDS pitches: 16B-aligned stores, bank-skewed

template <bool A_F32, bool W_F32, bool OUT_BF16, bool RMAP>
__global__ __launch_bounds__(256)
void gemm_nt(const void* __restrict__ Ap, int lda, long long sAb,
             int rmap_dim, int rmap_mul,
             const void* __restrict__ Wp, int ldw, long long sWb,
             void* __restrict__ Op, long long ldo, long long sOb,
             int M, int N, int K)
{
  __shared__ unsigned short At[2][BM * AP];
  __shared__ unsigned short Wt[2][BN * WP];

  const int tid  = threadIdx.x;
  const int lane = tid & 31;
  const int wave = tid >> 5;
  const int hx   = lane >> 4;          // lane half (0/1)
  const int lm   = lane & 15;
  const int wm0  = (wave >> 2) * 64;   // wave row origin in block tile
  const int wn0  = (wave & 3) * 32;    // wave col origin in block tile

  const int blockM = blockIdx.y * BM;
  const int blockN = blockIdx.x * BN;
  const int bz     = blockIdx.z;

  // --- staging geometry (straight-line): 16 elems/thread for each tile ---
  const int rA = tid >> 1, cA = (tid & 1) * 16;
  const int grA = blockM + rA;
  const int mrA = RMAP ? ((grA % rmap_dim) * rmap_mul + grA / rmap_dim) : grA;
  const size_t Abase = (size_t)bz * (size_t)sAb + (size_t)mrA * lda + cA;
  const size_t Wbase = (size_t)bz * (size_t)sWb + (size_t)(blockN + rA) * ldw + cA;
  const size_t ooff  = (size_t)bz * (size_t)sOb;

  // Two register stage-sets (distance-2 pipeline)
  St8<A_F32> a0_0, a1_0, a0_1, a1_1;
  St8<W_F32> w0_0, w1_0, w0_1, w1_1;

  auto load0 = [&](int kb) {
    a0_0.load(Ap, Abase + kb); a1_0.load(Ap, Abase + kb + 8);
    w0_0.load(Wp, Wbase + kb); w1_0.load(Wp, Wbase + kb + 8);
  };
  auto load1 = [&](int kb) {
    a0_1.load(Ap, Abase + kb); a1_1.load(Ap, Abase + kb + 8);
    w0_1.load(Wp, Wbase + kb); w1_1.load(Wp, Wbase + kb + 8);
  };
  auto store0 = [&](int p) {
    *(u16x8*)&At[p][rA * AP + cA]     = a0_0.pack();
    *(u16x8*)&At[p][rA * AP + cA + 8] = a1_0.pack();
    *(u16x8*)&Wt[p][rA * WP + cA]     = w0_0.pack();
    *(u16x8*)&Wt[p][rA * WP + cA + 8] = w1_0.pack();
  };
  auto store1 = [&](int p) {
    *(u16x8*)&At[p][rA * AP + cA]     = a0_1.pack();
    *(u16x8*)&At[p][rA * AP + cA + 8] = a1_1.pack();
    *(u16x8*)&Wt[p][rA * WP + cA]     = w0_1.pack();
    *(u16x8*)&Wt[p][rA * WP + cA + 8] = w1_1.pack();
  };

  v8f acc[4][2];
#pragma unroll
  for (int mt = 0; mt < 4; ++mt)
#pragma unroll
    for (int nt = 0; nt < 2; ++nt)
#pragma unroll
      for (int v = 0; v < 8; ++v) acc[mt][nt][v] = 0.0f;

  auto compute = [&](int p) {
    v16bf a[4], b[2];
#pragma unroll
    for (int mt = 0; mt < 4; ++mt) {
      FragCast u;
#pragma unroll
      for (int i = 0; i < 16; ++i) {
        int v = i >> 1, q = i & 1;
        // A 16x32 bf16: lanes<16 K in {0..7,16..23}, lanes>=16 K in {8..15,24..31}
        int k = ((v < 4) ? (v * 2 + q) : (16 + (v - 4) * 2 + q)) + hx * 8;
        u.s[i] = (short)At[p][(wm0 + mt * 16 + lm) * AP + k];
      }
      a[mt] = u.b;
    }
#pragma unroll
    for (int nt = 0; nt < 2; ++nt) {
      FragCast u;
#pragma unroll
      for (int i = 0; i < 16; ++i)  // B 32x16: N=lane%16, K same order as A
        u.s[i] = (short)Wt[p][(wn0 + nt * 16 + lm) * WP + ((i < 8) ? i : i + 8) + hx * 8];
      b[nt] = u.b;
    }
#pragma unroll
    for (int mt = 0; mt < 4; ++mt)
#pragma unroll
      for (int nt = 0; nt < 2; ++nt)
        acc[mt][nt] = __builtin_amdgcn_wmma_f32_16x16x32_bf16(
            false, a[mt], false, b[nt], (short)0, acc[mt][nt], false, false);
  };

  // --- distance-2 pipeline (K/BK >= 2, even, for all our shapes) ---
  // iter j: barrier; store block j+1 (loads issued at j-1); load block j+2;
  //         compute block j.
  load0(0);
  store0(0);          // prologue: one-time un-hidden wait
  load1(BK);
  int p = 0;
  int kb = 0;
  while (true) {
    // even j: block kb in buf p, set1 holds kb+BK, load into set0
    __syncthreads();
    if (kb + BK < K) store1(p ^ 1);
    if (kb + 2 * BK < K) load0(kb + 2 * BK);
    compute(p);
    p ^= 1; kb += BK;
    if (kb >= K) break;
    // odd j
    __syncthreads();
    if (kb + BK < K) store0(p ^ 1);
    if (kb + 2 * BK < K) load1(kb + 2 * BK);
    compute(p);
    p ^= 1; kb += BK;
    if (kb >= K) break;
  }

  // ---- epilogue: D layout M = v + half*8, N = lane%16 ----
#pragma unroll
  for (int mt = 0; mt < 4; ++mt)
#pragma unroll
    for (int nt = 0; nt < 2; ++nt)
#pragma unroll
      for (int v = 0; v < 8; ++v) {
        int row = blockM + wm0 + mt * 16 + hx * 8 + v;
        int col = blockN + wn0 + nt * 16 + lm;
        size_t o = ooff + (size_t)row * (size_t)ldo + col;
        if (OUT_BF16) ((unsigned short*)Op)[o] = f2bf(acc[mt][nt][v]);
        else          ((float*)Op)[o]          = acc[mt][nt][v];
      }
}

// ---------------------------------------------------------------------------
// Masking: where(mask) -> -inf, written into the final score layout
// ---------------------------------------------------------------------------
__global__ __launch_bounds__(256) void mask_ne(float* out, const unsigned char* nemask) {
  size_t total = (size_t)(BSZ * T_) * NE;
  for (size_t i = (size_t)blockIdx.x * blockDim.x + threadIdx.x; i < total;
       i += (size_t)gridDim.x * blockDim.x) {
    int r = (int)(i >> 13);            // / NE (8192)
    int n = (int)(i & (NE - 1));
    int b = r >> 8;                    // / T (256)
    if (nemask[((size_t)b << 13) + n])
      out[(size_t)r * ROWW + n] = NEGINF;
  }
}

__global__ __launch_bounds__(256) void mask_src(float* out, const int* src) {
  // total = 2048 * 256, launched exactly
  int i = blockIdx.x * 256 + threadIdx.x;
  int r = i >> 8;
  int sp = i & 255;
  int b = r >> 8;
  if (src[sp * BSZ + b] == 0)
    out[(size_t)r * ROWW + NE + VOCAB + sp] = NEGINF;
}

// ---------------------------------------------------------------------------
// Streaming log-softmax: branchless two-sweep per chunk, then combine
// ---------------------------------------------------------------------------
__device__ __forceinline__ void lse_combine(float& m, float& s, float m2, float s2) {
  if (m2 == NEGINF) return;
  if (m == NEGINF) { m = m2; s = s2; return; }
  if (m2 > m) { s = s * expf(m - m2) + s2; m = m2; }
  else        { s += s2 * expf(m2 - m); }
}

__device__ __forceinline__ void block_lse_reduce(float& m, float& s) {
  __shared__ float sm[256], ss[256];
  sm[threadIdx.x] = m; ss[threadIdx.x] = s;
  __syncthreads();
  for (int st = 128; st > 0; st >>= 1) {
    if ((int)threadIdx.x < st) {
      float mm = sm[threadIdx.x], sv = ss[threadIdx.x];
      lse_combine(mm, sv, sm[threadIdx.x + st], ss[threadIdx.x + st]);
      sm[threadIdx.x] = mm; ss[threadIdx.x] = sv;
    }
    __syncthreads();
  }
  m = sm[0]; s = ss[0];
}

__global__ __launch_bounds__(256) void reduce1(const float* __restrict__ out,
                                               float* pm, float* ps) {
  int b = blockIdx.y, cx = blockIdx.x;
  const float4* p4 = (const float4*)(out + (size_t)b * LBATCH + (size_t)cx * CHUNK);
  // Sweep 1: branchless max
  float m = NEGINF;
  for (int i = threadIdx.x; i < CHUNK / 4; i += 256) {
    float4 x = p4[i];
    m = fmaxf(m, fmaxf(fmaxf(x.x, x.y), fmaxf(x.z, x.w)));
  }
  // Sweep 2: branchless exp-sum (second read hits L2); exp(-inf - m) == 0
  float s = 0.f;
  if (m != NEGINF) {
    for (int i = threadIdx.x; i < CHUNK / 4; i += 256) {
      float4 x = p4[i];
      s += expf(x.x - m) + expf(x.y - m) + expf(x.z - m) + expf(x.w - m);
    }
  }
  block_lse_reduce(m, s);
  if (threadIdx.x == 0) { pm[b * NBLK + cx] = m; ps[b * NBLK + cx] = s; }
}

__global__ __launch_bounds__(256) void reduce2(const float* pm, const float* ps,
                                               float* lse) {
  int b = blockIdx.x;
  float m = NEGINF, s = 0.f;
  for (int i = threadIdx.x; i < NBLK; i += 256)
    lse_combine(m, s, pm[b * NBLK + i], ps[b * NBLK + i]);
  block_lse_reduce(m, s);
  if (threadIdx.x == 0) lse[b] = m + logf(s);
}

__global__ __launch_bounds__(256) void sub_lse(float* out, const float* lse) {
  int b = blockIdx.y;
  float v = lse[b];
  float4* p4 = (float4*)(out + (size_t)b * LBATCH);
  size_t n4 = (size_t)(LBATCH / 4);
  for (size_t i = (size_t)blockIdx.x * blockDim.x + threadIdx.x; i < n4;
       i += (size_t)gridDim.x * blockDim.x) {
    float4 x = p4[i];
    x.x -= v; x.y -= v; x.z -= v; x.w -= v;
    p4[i] = x;
  }
}

// ---------------------------------------------------------------------------
// Launch
// ---------------------------------------------------------------------------
extern "C" void kernel_launch(void* const* d_in, const int* in_sizes, int n_in,
                              void* d_out, int out_size, void* d_ws, size_t ws_size,
                              hipStream_t stream) {
  const float* enccanv = (const float*)d_in[0];   // (256, 8, 512)
  const float* encne   = (const float*)d_in[1];   // (64, 128, 512) -> (8192, 512)
  const float* encsrc  = (const float*)d_in[2];   // (256, 8, 512)  -> (2048, 512)
  const float* outlut  = (const float*)d_in[3];   // (32000, 512)
  const float* C1w     = (const float*)d_in[4];   // (512, 512)
  const float* N1w     = (const float*)d_in[5];
  const float* S1w     = (const float*)d_in[6];
  const float* W1w     = (const float*)d_in[7];
  const unsigned char* nemask = (const unsigned char*)d_in[8];  // (8, 8192) bool
  const int* src       = (const int*)d_in[9];     // (256, 8)
  float* out = (float*)d_out;                     // (2048, 40448) row-major

  // Workspace: bf16 (ushort) intermediates + reduction partials (~45 MB)
  unsigned short* Cb = (unsigned short*)d_ws;            // 2048 x 512
  unsigned short* nk = Cb + (size_t)2048 * 512;          // 8192 x 512
  unsigned short* sk = nk + (size_t)NE * 512;            // 2048 x 512
  unsigned short* wk = sk + (size_t)2048 * 512;          // 32000 x 512
  float* pm  = (float*)(wk + (size_t)VOCAB * 512);
  float* ps  = pm + BSZ * NBLK;
  float* lse = ps + BSZ * NBLK;

  dim3 blk(256);

  // --- projections (f32 in, bf16 out), y = x @ W.T, K=512 ---
  gemm_nt<true, true, true, true><<<dim3(4, 16, 1), blk, 0, stream>>>(
      enccanv, DIM, 0, /*rmap*/ T_, BSZ, C1w, DIM, 0, Cb, DIM, 0, 2048, DIM, DIM);
  gemm_nt<true, true, true, false><<<dim3(4, 64, 1), blk, 0, stream>>>(
      encne, DIM, 0, 0, 0, N1w, DIM, 0, nk, DIM, 0, NE, DIM, DIM);
  gemm_nt<true, true, true, false><<<dim3(4, 16, 1), blk, 0, stream>>>(
      encsrc, DIM, 0, 0, 0, S1w, DIM, 0, sk, DIM, 0, 2048, DIM, DIM);
  gemm_nt<true, true, true, false><<<dim3(4, 250, 1), blk, 0, stream>>>(
      outlut, DIM, 0, 0, 0, W1w, DIM, 0, wk, DIM, 0, VOCAB, DIM, DIM);

  // --- score GEMMs written directly into concatenated layout (ldo=40448) ---
  gemm_nt<false, false, false, false><<<dim3(NE / BN, 16, 1), blk, 0, stream>>>(
      Cb, DIM, 0, 0, 0, nk, DIM, 0, out, ROWW, 0, 2048, NE, DIM);
  gemm_nt<false, false, false, false><<<dim3(VOCAB / BN, 16, 1), blk, 0, stream>>>(
      Cb, DIM, 0, 0, 0, wk, DIM, 0, out + NE, ROWW, 0, 2048, VOCAB, DIM);
  // sscores: batched over bsz; W rows are skeys[s, b, :] (ldw = bsz*dim)
  gemm_nt<false, false, false, false><<<dim3(SRCL / BN, T_ / BM, BSZ), blk, 0, stream>>>(
      Cb, DIM, (long long)T_ * DIM, 0, 0,
      sk, BSZ * DIM, (long long)DIM,
      out + NE + VOCAB, ROWW, (long long)T_ * ROWW, T_, SRCL, DIM);

  // --- masks -> -inf ---
  mask_ne<<<dim3(4096), blk, 0, stream>>>(out, nemask);
  mask_src<<<dim3(2048), blk, 0, stream>>>(out, src);

  // --- log-softmax over each batch's 10.35M scores ---
  reduce1<<<dim3(NBLK, BSZ), blk, 0, stream>>>(out, pm, ps);
  reduce2<<<dim3(BSZ), blk, 0, stream>>>(pm, ps, lse);
  sub_lse<<<dim3(2048, BSZ), blk, 0, stream>>>(out, lse);
}
